// Attention_10892037063349
// MI455X (gfx1250) — matrix operand order repaired and checked by reading, not compiled, not run on hardware
//
#include <hip/hip_runtime.h>
#include <math.h>

// ---------------- problem constants ----------------
#define Bsz  2
#define Ssz  2048
#define Dsz  2048
#define Hn   16
#define KVn  4
#define HDsz 128
#define NREP 4
#define Rr   16
#define Msz  (Bsz * Ssz)          // 4096
#define BK   32                   // WMMA contraction step (bf16 K=32)

// gfx1250 async global->LDS DMA path (ASYNCcnt-tracked), with graceful fallback
#if defined(__has_builtin)
#  if __has_builtin(__builtin_amdgcn_global_load_async_to_lds_b128)
#    define USE_ASYNC_LDS 1
#  endif
#endif
#ifndef USE_ASYNC_LDS
#  define USE_ASYNC_LDS 0
#endif

typedef __attribute__((ext_vector_type(16))) __bf16 v16bf;
typedef __attribute__((ext_vector_type(8)))  __bf16 v8bf;
typedef __attribute__((ext_vector_type(8)))  float  v8f;
typedef __attribute__((ext_vector_type(4)))  int    v4i;

#if USE_ASYNC_LDS
// async builtin signature: (v4i addrspace(1)* gsrc, v4i addrspace(3)* ldst, imm off, imm cpol)
typedef __attribute__((address_space(1))) v4i gv4i;
typedef __attribute__((address_space(3))) v4i lv4i;
#endif

union Frag {
  v16bf v;
  v8bf  h[2];
};

__device__ __forceinline__ unsigned short f2bf(float f) {
  union { float f; unsigned u; } x; x.f = f;
  unsigned r = x.u + 0x7FFFu + ((x.u >> 16) & 1u);
  return (unsigned short)(r >> 16);
}

__device__ __forceinline__ v8f wmma_bf16(v16bf a, v16bf b, v8f c) {
  // D = A(16x32 bf16) * B(32x16 bf16) + C(16x16 f32)
  return __builtin_amdgcn_wmma_f32_16x16x32_bf16(
      /*neg_a=*/false, a, /*neg_b=*/false, b,
      /*c_mod=*/(short)0, c, /*reuse_a=*/false, /*reuse_b=*/false);
}

// ---------------- fp32 -> bf16 convert ----------------
__global__ __launch_bounds__(256) void k_f2bf(const float* __restrict__ src,
                                              unsigned short* __restrict__ dst,
                                              int n) {
  int i = blockIdx.x * 256 + threadIdx.x;
  if (i < n) dst[i] = f2bf(src[i]);
}

// ---------------- fold LoRA into weight: We = bf16(W + s * Bm@A) ----------------
__global__ __launch_bounds__(256) void k_fuse_lora(const float* __restrict__ w,
                                                   const float* __restrict__ a,
                                                   const float* __restrict__ bm,
                                                   unsigned short* __restrict__ we,
                                                   int N, int D, float scaling) {
  int i = blockIdx.x * 256 + threadIdx.x;
  if (i >= N * D) return;
  int d = i % D, n = i / D;
  float s = 0.f;
#pragma unroll
  for (int r = 0; r < Rr; ++r) s += bm[n * Rr + r] * a[r * D + d];
  we[i] = f2bf(w[i] + scaling * s);
}

// ---------------- bf16 GEMM: C[M,N] = A[M,K] * Bw[N,K]^T ----------------
// 256 threads = 8 waves; block tile 128x128; wave tile 64x32 (4x2 wmma accums)
// Double-buffered LDS fed by GLOBAL_LOAD_ASYNC_TO_LDS (4 async issues per wave
// per K-tile => s_wait_asynccnt 4 overlaps next tile's DMA with current WMMAs).
__global__ __launch_bounds__(256) void k_gemm_bf16(
    const unsigned short* __restrict__ A,   // [M,K] bf16 row-major
    const unsigned short* __restrict__ Bw,  // [N,K] bf16 row-major
    float* __restrict__ C, int M, int N, int K) {
  __shared__ unsigned short As[2][128][BK + 8];
  __shared__ unsigned short Bs[2][128][BK + 8];
  const int tid  = threadIdx.x;
  const int lane = tid & 31;
  const int wave = tid >> 5;
  const int hf   = lane >> 4;       // half-wave id
  const int l16  = lane & 15;
  const int bm0  = blockIdx.x * 128;
  const int bn0  = blockIdx.y * 128;
  const int wm0  = (wave >> 2) * 64;
  const int wn0  = (wave & 3) * 32;

  v8f acc[4][2];
  for (int i = 0; i < 4; ++i)
    for (int j = 0; j < 2; ++j)
      for (int e = 0; e < 8; ++e) acc[i][j][e] = 0.f;

  const int ra = tid >> 2;          // 0..63 (row per pass)
  const int ca = (tid & 3) * 8;     // 0,8,16,24
  const int nT = K / BK;

  auto issue_tile = [&](int t, int buf) {
    const int k0 = t * BK;
#pragma unroll
    for (int p = 0; p < 2; ++p) {
      const int r = p * 64 + ra;
      const unsigned short* ga = &A[(size_t)(bm0 + r) * K + k0 + ca];
      const unsigned short* gb = &Bw[(size_t)(bn0 + r) * K + k0 + ca];
#if USE_ASYNC_LDS
      __builtin_amdgcn_global_load_async_to_lds_b128(
          (gv4i*)ga, (lv4i*)&As[buf][r][ca], 0, 0);
      __builtin_amdgcn_global_load_async_to_lds_b128(
          (gv4i*)gb, (lv4i*)&Bs[buf][r][ca], 0, 0);
#else
      *reinterpret_cast<v8bf*>(&As[buf][r][ca]) = *reinterpret_cast<const v8bf*>(ga);
      *reinterpret_cast<v8bf*>(&Bs[buf][r][ca]) = *reinterpret_cast<const v8bf*>(gb);
#endif
    }
  };

  issue_tile(0, 0);                   // prologue: DMA tile 0 into buffer 0

  for (int t = 0; t < nT; ++t) {
    const int buf = t & 1;
    if (t + 1 < nT) {
      issue_tile(t + 1, (t + 1) & 1); // overlap next tile's DMA with this tile's WMMAs
#if USE_ASYNC_LDS
      asm volatile("s_wait_asynccnt 4" ::: "memory");  // tile t landed; t+1 in flight
#endif
    } else {
#if USE_ASYNC_LDS
      asm volatile("s_wait_asynccnt 0" ::: "memory");
#endif
    }
    __syncthreads();

    // A fragment: lanes 0-15 -> K {0..7,16..23}, lanes 16-31 -> K {8..15,24..31}
    const int ac0 = hf ? 8 : 0;
    // B fragment: lanes 0-15 -> K {0..7,16..23}, lanes 16-31 -> K {8..15,24..31}
    const int bc0 = hf ? 8 : 0;
    Frag af[4], bfr[2];
#pragma unroll
    for (int i = 0; i < 4; ++i) {
      int row = wm0 + i * 16 + l16;
      af[i].h[0] = *reinterpret_cast<const v8bf*>(&As[buf][row][ac0]);
      af[i].h[1] = *reinterpret_cast<const v8bf*>(&As[buf][row][ac0 + 16]);
    }
#pragma unroll
    for (int j = 0; j < 2; ++j) {
      int col = wn0 + j * 16 + l16;
      bfr[j].h[0] = *reinterpret_cast<const v8bf*>(&Bs[buf][col][bc0]);
      bfr[j].h[1] = *reinterpret_cast<const v8bf*>(&Bs[buf][col][bc0 + 16]);
    }
#pragma unroll
    for (int i = 0; i < 4; ++i)
#pragma unroll
      for (int j = 0; j < 2; ++j)
        acc[i][j] = wmma_bf16(af[i].v, bfr[j].v, acc[i][j]);
    __syncthreads();                  // everyone done reading buf before its reuse
  }

  // C/D layout: lanes 0-15: M=r, N=lane; lanes 16-31: M=8+r, N=lane-16
#pragma unroll
  for (int i = 0; i < 4; ++i)
#pragma unroll
    for (int j = 0; j < 2; ++j)
#pragma unroll
      for (int r = 0; r < 8; ++r) {
        int row = bm0 + wm0 + i * 16 + hf * 8 + r;
        int col = bn0 + wn0 + j * 16 + l16;
        C[(size_t)row * N + col] = acc[i][j][r];
      }
}

// ---------------- RoPE on Q, scale by 1/sqrt(HD), to [B,H,S,HD] bf16 ----------------
__global__ __launch_bounds__(256) void k_rope_q(const float* __restrict__ q,
                                                const float* __restrict__ freqs,
                                                unsigned short* __restrict__ qb) {
  int idx = blockIdx.x * 256 + threadIdx.x;
  if (idx >= Bsz * Ssz * Hn * (HDsz / 2)) return;
  const int i = idx & 63; int t = idx >> 6;
  const int h = t & (Hn - 1); t >>= 4;
  const int s = t & (Ssz - 1);
  const int b = t >> 11;
  const float f = freqs[s * (HDsz / 2) + i];
  const float c = cosf(f), sn = sinf(f);
  const size_t src = ((size_t)(b * Ssz + s)) * (Hn * HDsz) + h * HDsz + 2 * i;
  const float x0 = q[src], x1 = q[src + 1];
  const float scale = 0.08838834764831845f;  // 1/sqrt(128)
  const size_t dst = ((size_t)(b * Hn + h) * Ssz + s) * HDsz + 2 * i;
  qb[dst]     = f2bf((x0 * c - x1 * sn) * scale);
  qb[dst + 1] = f2bf((x0 * sn + x1 * c) * scale);
}

// ---------------- RoPE on K -> kb bf16 [B,KV,S,HD] + kh fp32 [B,H,S,HD] (GQA expand) ----
__global__ __launch_bounds__(256) void k_rope_k(const float* __restrict__ k,
                                                const float* __restrict__ freqs,
                                                unsigned short* __restrict__ kb,
                                                float* __restrict__ kh) {
  int idx = blockIdx.x * 256 + threadIdx.x;
  if (idx >= Bsz * Ssz * KVn * (HDsz / 2)) return;
  const int i = idx & 63; int t = idx >> 6;
  const int kv = t & (KVn - 1); t >>= 2;
  const int s = t & (Ssz - 1);
  const int b = t >> 11;
  const float f = freqs[s * (HDsz / 2) + i];
  const float c = cosf(f), sn = sinf(f);
  const size_t src = ((size_t)(b * Ssz + s)) * (KVn * HDsz) + kv * HDsz + 2 * i;
  const float x0 = k[src], x1 = k[src + 1];
  const float r0 = x0 * c - x1 * sn, r1 = x0 * sn + x1 * c;
  const size_t dk = ((size_t)(b * KVn + kv) * Ssz + s) * HDsz + 2 * i;
  kb[dk] = f2bf(r0); kb[dk + 1] = f2bf(r1);
#pragma unroll
  for (int j = 0; j < NREP; ++j) {
    const size_t o = ((size_t)(b * Hn + kv * NREP + j) * Ssz + s) * HDsz + 2 * i;
    kh[o] = r0; kh[o + 1] = r1;
  }
}

// ---------------- V -> vbT bf16 [B,KV,HD,S] (transposed) + vh fp32 expand ----------------
__global__ __launch_bounds__(256) void k_vexp(const float* __restrict__ v,
                                              unsigned short* __restrict__ vbT,
                                              float* __restrict__ vh) {
  int idx = blockIdx.x * 256 + threadIdx.x;
  if (idx >= Bsz * Ssz * KVn * HDsz) return;
  const int d = idx & (HDsz - 1); int t = idx >> 7;
  const int kv = t & (KVn - 1); t >>= 2;
  const int s = t & (Ssz - 1);
  const int b = t >> 11;
  const float val = v[((size_t)(b * Ssz + s)) * (KVn * HDsz) + kv * HDsz + d];
  vbT[((size_t)(b * KVn + kv) * HDsz + d) * Ssz + s] = f2bf(val);
#pragma unroll
  for (int j = 0; j < NREP; ++j)
    vh[((size_t)(b * Hn + kv * NREP + j) * Ssz + s) * HDsz + d] = val;
}

// ---------------- flash attention: per wave 16 q-rows x HD=128 ----------------
__global__ __launch_bounds__(256) void k_flash(const unsigned short* __restrict__ qb,
                                               const unsigned short* __restrict__ kb,
                                               const unsigned short* __restrict__ vbT,
                                               unsigned short* __restrict__ attnb) {
  __shared__ unsigned short Pl[8][16][BK + 8];   // per-wave P tile (bf16)
  const int tid  = threadIdx.x;
  const int lane = tid & 31;
  const int wave = tid >> 5;
  const int hf   = lane >> 4;
  const int l16  = lane & 15;
  const int bh = blockIdx.x;
  const int b  = bh / Hn;
  const int h  = bh % Hn;
  const int kv = h / NREP;
  const int q0 = blockIdx.y * 128 + wave * 16;
  const size_t qbase = ((size_t)(b * Hn + h) * Ssz + q0) * HDsz;
  const size_t kbase = (size_t)(b * KVn + kv) * Ssz * HDsz;
  const size_t vbase = (size_t)(b * KVn + kv) * HDsz * Ssz;
  const int c0 = hf ? 8 : 0;

  // preload Q as 4 A-fragments (HD=128 = 4 * K32), already scaled by 1/sqrt(HD)
  Frag qf4[4];
#pragma unroll
  for (int ks = 0; ks < 4; ++ks) {
    const unsigned short* p = &qb[qbase + (size_t)l16 * HDsz + ks * 32 + c0];
    qf4[ks].h[0] = *reinterpret_cast<const v8bf*>(p);
    qf4[ks].h[1] = *reinterpret_cast<const v8bf*>(p + 16);
  }

  v8f o[8];
  for (int nt = 0; nt < 8; ++nt)
    for (int e = 0; e < 8; ++e) o[nt][e] = 0.f;
  float mrun[8], lrun[8];
  for (int r = 0; r < 8; ++r) { mrun[r] = -1e30f; lrun[r] = 0.f; }

  const int kvEnd = blockIdx.y * 128 + 128;  // causal bound for this q-block
  for (int kt = 0; kt < kvEnd; kt += 32) {
    // ---- scores S = Q * K^T : 16x32 (two 16x16 wmma tiles, 4 K-steps each) ----
    v8f sc[2];
    for (int nt = 0; nt < 2; ++nt)
      for (int e = 0; e < 8; ++e) sc[nt][e] = 0.f;
#pragma unroll
    for (int nt = 0; nt < 2; ++nt) {
#pragma unroll
      for (int ks = 0; ks < 4; ++ks) {
        Frag kf;
        const unsigned short* p =
            &kb[kbase + (size_t)(kt + nt * 16 + l16) * HDsz + ks * 32 + c0];
        kf.h[0] = *reinterpret_cast<const v8bf*>(p);
        kf.h[1] = *reinterpret_cast<const v8bf*>(p + 16);
        sc[nt] = wmma_bf16(qf4[ks].v, kf.v, sc[nt]);
      }
    }
    // ---- causal mask + online softmax (row stats shared by 16-lane groups) ----
#pragma unroll
    for (int r = 0; r < 8; ++r) {
      const int row = q0 + hf * 8 + r;
      float mx = -1e30f;
#pragma unroll
      for (int nt = 0; nt < 2; ++nt) {
        const int col = kt + nt * 16 + l16;
        float s = sc[nt][r];
        if (col > row) s = -1e30f;
        sc[nt][r] = s;
        mx = fmaxf(mx, s);
      }
#pragma unroll
      for (int m = 1; m < 16; m <<= 1) mx = fmaxf(mx, __shfl_xor(mx, m, 32));
      const float mnew = fmaxf(mrun[r], mx);
      const float corr = __expf(mrun[r] - mnew);
      mrun[r] = mnew;
      lrun[r] *= corr;
#pragma unroll
      for (int nt = 0; nt < 8; ++nt) o[nt][r] *= corr;
      float rs = 0.f;
#pragma unroll
      for (int nt = 0; nt < 2; ++nt) {
        const float p = __expf(sc[nt][r] - mnew);
        sc[nt][r] = p;
        rs += p;
      }
#pragma unroll
      for (int m = 1; m < 16; m <<= 1) rs += __shfl_xor(rs, m, 32);
      lrun[r] += rs;
    }
    // ---- P (f32 C-layout) -> bf16 A-fragment via per-wave LDS round trip ----
#pragma unroll
    for (int nt = 0; nt < 2; ++nt)
#pragma unroll
      for (int r = 0; r < 8; ++r)
        Pl[wave][hf * 8 + r][nt * 16 + l16] = f2bf(sc[nt][r]);
    asm volatile("s_wait_dscnt 0" ::: "memory");   // LDS per-wave in-order + fence
    Frag pf;
    pf.h[0] = *reinterpret_cast<const v8bf*>(&Pl[wave][l16][c0]);
    pf.h[1] = *reinterpret_cast<const v8bf*>(&Pl[wave][l16][c0 + 16]);
    // ---- O += P * V : 8 wmma tiles across HD (V^T layout -> contiguous K) ----
#pragma unroll
    for (int nt = 0; nt < 8; ++nt) {
      Frag vf;
      const unsigned short* p =
          &vbT[vbase + (size_t)(nt * 16 + l16) * Ssz + kt + c0];
      vf.h[0] = *reinterpret_cast<const v8bf*>(p);
      vf.h[1] = *reinterpret_cast<const v8bf*>(p + 16);
      o[nt] = wmma_bf16(pf.v, vf.v, o[nt]);
    }
    asm volatile("s_wait_dscnt 0" ::: "memory");   // WAR fence before next P store
  }
  // ---- finalize: O / l, write bf16 attn [B,S,H*HD] ----
#pragma unroll
  for (int nt = 0; nt < 8; ++nt)
#pragma unroll
    for (int r = 0; r < 8; ++r) {
      const int row = q0 + hf * 8 + r;
      const float inv = 1.f / lrun[r];
      attnb[((size_t)(b * Ssz + row)) * (Hn * HDsz) + h * HDsz + nt * 16 + l16] =
          f2bf(o[nt][r] * inv);
    }
}

// ---------------- workspace layout (bytes) ----------------
static constexpr size_t SZ_XB  = (size_t)Msz * Dsz * 2;            // bf16 x (reused as attn)
static constexpr size_t SZ_WQE = (size_t)Dsz * Dsz * 2;
static constexpr size_t SZ_WKE = (size_t)(KVn * HDsz) * Dsz * 2;
static constexpr size_t SZ_WVE = SZ_WKE;
static constexpr size_t SZ_WOB = (size_t)Dsz * Dsz * 2;
static constexpr size_t SZ_QF  = (size_t)Msz * Dsz * 4;            // fp32 gemm out scratch
static constexpr size_t SZ_QB  = (size_t)Bsz * Hn * Ssz * HDsz * 2;
static constexpr size_t SZ_KB  = (size_t)Bsz * KVn * Ssz * HDsz * 2;

static constexpr size_t OFF_XB  = 0;
static constexpr size_t OFF_WQE = OFF_XB + SZ_XB;
static constexpr size_t OFF_WKE = OFF_WQE + SZ_WQE;
static constexpr size_t OFF_WVE = OFF_WKE + SZ_WKE;
static constexpr size_t OFF_WOB = OFF_WVE + SZ_WVE;
static constexpr size_t OFF_QF  = OFF_WOB + SZ_WOB;
static constexpr size_t OFF_QB  = OFF_QF + SZ_QF;
static constexpr size_t OFF_KB  = OFF_QB + SZ_QB;
static constexpr size_t OFF_VBT = OFF_KB + SZ_KB;

extern "C" void kernel_launch(void* const* d_in, const int* in_sizes, int n_in,
                              void* d_out, int out_size, void* d_ws, size_t ws_size,
                              hipStream_t stream) {
  (void)in_sizes; (void)n_in; (void)out_size; (void)ws_size;
  const float* x     = (const float*)d_in[0];
  const float* freqs = (const float*)d_in[1];
  // d_in[2] = mask (unused: causal mask applied analytically)
  const float* wq   = (const float*)d_in[3];
  const float* wq_a = (const float*)d_in[4];
  const float* wq_b = (const float*)d_in[5];
  const float* wk   = (const float*)d_in[6];
  const float* wk_a = (const float*)d_in[7];
  const float* wk_b = (const float*)d_in[8];
  const float* wv   = (const float*)d_in[9];
  const float* wv_a = (const float*)d_in[10];
  const float* wv_b = (const float*)d_in[11];
  const float* wo   = (const float*)d_in[12];

  char* ws = (char*)d_ws;
  unsigned short* xb    = (unsigned short*)(ws + OFF_XB);
  unsigned short* wqe   = (unsigned short*)(ws + OFF_WQE);
  unsigned short* wke   = (unsigned short*)(ws + OFF_WKE);
  unsigned short* wve   = (unsigned short*)(ws + OFF_WVE);
  unsigned short* wob   = (unsigned short*)(ws + OFF_WOB);
  float*          qf    = (float*)(ws + OFF_QF);
  unsigned short* qb    = (unsigned short*)(ws + OFF_QB);
  unsigned short* kb    = (unsigned short*)(ws + OFF_KB);
  unsigned short* vbT   = (unsigned short*)(ws + OFF_VBT);
  unsigned short* attnb = xb;   // xb dead after QKV gemms; reuse for attention out

  float* out = (float*)d_out;
  float* kh  = out + (size_t)Bsz * Ssz * Dsz;             // 8,388,608
  float* vh  = kh + (size_t)Bsz * Hn * Ssz * HDsz;        // +8,388,608

  const float SCALING = 16.0f / (float)Rr;  // 1.0
  dim3 blk(256);

  // 1) convert x and wo to bf16; fold LoRA into Q/K/V weights (bf16)
  k_f2bf<<<(Msz * Dsz + 255) / 256, blk, 0, stream>>>(x, xb, Msz * Dsz);
  k_f2bf<<<(Dsz * Dsz + 255) / 256, blk, 0, stream>>>(wo, wob, Dsz * Dsz);
  k_fuse_lora<<<(Dsz * Dsz + 255) / 256, blk, 0, stream>>>(wq, wq_a, wq_b, wqe, Dsz, Dsz, SCALING);
  k_fuse_lora<<<(KVn * HDsz * Dsz + 255) / 256, blk, 0, stream>>>(wk, wk_a, wk_b, wke, KVn * HDsz, Dsz, SCALING);
  k_fuse_lora<<<(KVn * HDsz * Dsz + 255) / 256, blk, 0, stream>>>(wv, wv_a, wv_b, wve, KVn * HDsz, Dsz, SCALING);

  // 2) Q = x @ Wq^T -> rope -> qb (bf16, pre-scaled)
  k_gemm_bf16<<<dim3(Msz / 128, Dsz / 128), blk, 0, stream>>>(xb, wqe, qf, Msz, Dsz, Dsz);
  k_rope_q<<<(Bsz * Ssz * Hn * 64 + 255) / 256, blk, 0, stream>>>(qf, freqs, qb);

  // 3) K = x @ Wk^T -> rope -> kb (bf16) + kh (fp32, GQA-expanded output)
  k_gemm_bf16<<<dim3(Msz / 128, (KVn * HDsz) / 128), blk, 0, stream>>>(xb, wke, qf, Msz, KVn * HDsz, Dsz);
  k_rope_k<<<(Bsz * Ssz * KVn * 64 + 255) / 256, blk, 0, stream>>>(qf, freqs, kb, kh);

  // 4) V = x @ Wv^T -> vbT (bf16 transposed) + vh (fp32, GQA-expanded output)
  k_gemm_bf16<<<dim3(Msz / 128, (KVn * HDsz) / 128), blk, 0, stream>>>(xb, wve, qf, Msz, KVn * HDsz, Dsz);
  k_vexp<<<(Bsz * Ssz * KVn * HDsz + 255) / 256, blk, 0, stream>>>(qf, vbT, vh);

  // 5) flash attention -> attnb (bf16, [B,S,H*HD])
  k_flash<<<dim3(Bsz * Hn, Ssz / 128), blk, 0, stream>>>(qb, kb, vbT, attnb);

  // 6) out = attn @ Wo^T (fp32 -> d_out)
  k_gemm_bf16<<<dim3(Msz / 128, Dsz / 128), blk, 0, stream>>>(attnb, wob, out, Msz, Dsz, Dsz);
}
